// ElemPatchContrastModel_75909251989856
// MI455X (gfx1250) — compile-verified, matrix operand order checked by reading, not run on hardware
//
#include <hip/hip_runtime.h>
#include <hip/hip_fp16.h>

typedef __attribute__((ext_vector_type(16))) _Float16 v16h;
typedef __attribute__((ext_vector_type(8)))  float    v8f;
typedef __attribute__((ext_vector_type(4)))  int      v4i;

#define B_    8
#define C_IN  6
#define N_    16384
#define D_    128
#define DP_   128
#define P_    64
#define K_    64
#define MAXMEM 1024

#if __has_builtin(__builtin_amdgcn_global_load_async_to_lds_b128)
#define USE_ASYNC_LDS 1
typedef __attribute__((address_space(1))) v4i gv4i;   // global int4 vector
typedef __attribute__((address_space(3))) v4i lv4i;   // LDS int4 vector
#else
#define USE_ASYNC_LDS 0
#endif

// ---------------------------------------------------------------------------
// Lane-local fragment loader for V_WMMA_F32_16X16X32_F16 operands.
// base points at a row-major [16][stride] half array where the 16-wide dim is
// M (for A) or N (for B). CDNA5 16-bit 16x32 layout: lane l -> row l%16,
// half-wave h=l>>4 selects K octet; VGPR q holds K = (q>=4?16:0)+8h+2(q&3)+{0,1}.
// All reads are dword-aligned (stride=128, k0 multiple of 32).
// ---------------------------------------------------------------------------
__device__ __forceinline__ v16h load_frag(const _Float16* base, int stride,
                                          int k0, int lane) {
  const int r = lane & 15;
  const int h = lane >> 4;
  const _Float16* row = base + r * stride + k0;
  union { v16h v; unsigned u[8]; } f;
#pragma unroll
  for (int q = 0; q < 8; ++q) {
    int k = ((q >> 2) << 4) + (h << 3) + ((q & 3) << 1);
    f.u[q] = *(const unsigned*)(row + k);
  }
  return f.v;
}

// Deterministic counter-hash Gumbel noise (stands in for jax threefry key 42).
__device__ __forceinline__ float gumbel_hash(int b, int p, int n) {
  unsigned x = (unsigned)((b * P_ + p) * N_ + n);
  x ^= 42u * 0x9E3779B9u;
  x ^= x >> 16; x *= 0x7FEB352Du;
  x ^= x >> 15; x *= 0x846CA68Bu;
  x ^= x >> 16;
  float u = (float)(x >> 8) * (1.0f / 16777216.0f);
  u = fminf(fmaxf(u, 1e-7f), 1.0f - 1e-7f);
  return -__logf(-__logf(u));
}

// ---------------------------------------------------------------------------
// K0: zero scratch + samples output region, convert w2 -> f16
// ---------------------------------------------------------------------------
__global__ void k_init(float* __restrict__ pooled, int* __restrict__ cnts,
                       float* __restrict__ samples,
                       const float* __restrict__ w2, _Float16* __restrict__ w2h) {
  const int i = blockIdx.x * blockDim.x + threadIdx.x;
  const int stride = gridDim.x * blockDim.x;
  for (int j = i; j < B_ * P_ * D_; j += stride) pooled[j] = 0.0f;
  for (int j = i; j < B_ * P_; j += stride) cnts[j] = 0;
  for (size_t j = i; j < (size_t)B_ * P_ * DP_ * K_; j += stride) samples[j] = 0.0f;
  for (int j = i; j < DP_ * D_; j += stride) w2h[j] = (_Float16)w2[j];
}

// ---------------------------------------------------------------------------
// K1: feats = relu(w1*features + b1), stored f16 as [b][n][d] (column of the
// math matrix contiguous), plus LDS segment-sum pooling per patch.
// ---------------------------------------------------------------------------
__global__ void k_feats_pool(const float* __restrict__ features,
                             const int* __restrict__ labels,
                             const float* __restrict__ w1,
                             const float* __restrict__ b1,
                             _Float16* __restrict__ feats16,
                             float* __restrict__ pooled,
                             int* __restrict__ cnts) {
  __shared__ float poolLoc[P_ * D_];  // 32 KB
  __shared__ int   cntLoc[P_];
  __shared__ float w1s[D_ * C_IN];
  __shared__ float b1s[D_];
  const int tid = threadIdx.x;
  const int NB = 128;
  const int b  = blockIdx.x / (N_ / NB);
  const int n0 = (blockIdx.x % (N_ / NB)) * NB;

  for (int i = tid; i < P_ * D_; i += 256) poolLoc[i] = 0.0f;
  for (int i = tid; i < P_; i += 256) cntLoc[i] = 0;
  for (int i = tid; i < D_ * C_IN; i += 256) w1s[i] = w1[i];
  for (int i = tid; i < D_; i += 256) b1s[i] = b1[i];
  __syncthreads();

  for (int it = 0; it < NB * D_ / 256; ++it) {
    int lin = it * 256 + tid;
    int nn = lin >> 7;       // 0..127
    int d  = lin & 127;
    int n  = n0 + nn;
    int lab = labels[b * N_ + n];
    float acc = b1s[d];
#pragma unroll
    for (int c = 0; c < C_IN; ++c)
      acc = fmaf(w1s[d * C_IN + c], features[((size_t)b * C_IN + c) * N_ + n], acc);
    acc = fmaxf(acc, 0.0f);
    feats16[((size_t)b * N_ + n) * D_ + d] = (_Float16)acc;
    atomicAdd(&poolLoc[lab * D_ + d], acc);
    if (d == 0) atomicAdd(&cntLoc[lab], 1);
  }
  __syncthreads();

  for (int i = tid; i < P_ * D_; i += 256) {
    float v = poolLoc[i];
    if (v != 0.0f) atomicAdd(&pooled[b * P_ * D_ + i], v);
  }
  for (int i = tid; i < P_; i += 256) {
    int c = cntLoc[i];
    if (c) atomicAdd(&cnts[b * P_ + i], c);
  }
}

// ---------------------------------------------------------------------------
// K2: patch_emb[bp][e] = w2 * mean(pooled) + b2 via WMMA.
// A = w2 (M = e), B = mean (N = bp): each lane's 8 results are contiguous in
// e, so the epilogue is two b128 stores per lane.
// Block = 16 bp rows, 8 waves = 8 e-tiles. K-loop 128/32 = 4 WMMAs.
// ---------------------------------------------------------------------------
__global__ void k_patch_emb(const float* __restrict__ pooled,
                            const int* __restrict__ cnts,
                            const _Float16* __restrict__ w2h,
                            const float* __restrict__ b2,
                            float* __restrict__ outEmb) {
  __shared__ _Float16 meanB[16 * D_];  // 4 KB
  const int tid  = threadIdx.x;
  const int lane = tid & 31;
  const int wave = tid >> 5;           // e-tile
  const int bp0  = blockIdx.x * 16;
  {
    int r = tid >> 4;                  // 0..15 bp row
    int part = tid & 15;               // 8 d's each
    int bp = bp0 + r;
    float c = (float)cnts[bp];
    float inv = (c != 0.0f) ? 1.0f / c : 1.0f;
#pragma unroll
    for (int j = 0; j < 8; ++j) {
      int d = part * 8 + j;
      meanB[r * D_ + d] = (_Float16)(pooled[bp * D_ + d] * inv);
    }
  }
  __syncthreads();

  const int e0 = wave * 16;
  v8f acc = {};
#pragma unroll
  for (int kk = 0; kk < 4; ++kk) {
    v16h a  = load_frag(w2h + e0 * D_, D_, kk * 32, lane);  // A: M=e,  K=d
    v16h bm = load_frag(&meanB[0],     D_, kk * 32, lane);  // B: N=bp, K=d
    acc = __builtin_amdgcn_wmma_f32_16x16x32_f16(false, a, false, bm,
                                                 (short)0, acc, false, false);
  }
  // D[m=e][n=bp]: lane -> bp col = lane%16; rows m = r + 8*(lane>>4).
  const int nb    = lane & 15;
  const int hh    = lane >> 4;
  const int ebase = e0 + 8 * hh;
  float vals[8];
#pragma unroll
  for (int r = 0; r < 8; ++r) vals[r] = acc[r] + b2[ebase + r];
  float* outp = outEmb + (bp0 + nb) * DP_ + ebase;   // 32B-aligned
  *(float4*)outp       = make_float4(vals[0], vals[1], vals[2], vals[3]);
  *((float4*)outp + 1) = make_float4(vals[4], vals[5], vals[6], vals[7]);
}

// ---------------------------------------------------------------------------
// K3: per (b,p) Gumbel top-K member selection. Members appended to LDS,
// 64 argmax rounds (selection removes winner by setting score to -inf).
// Also emits counts (int32) and sizes = floor(count/2) clamped to K.
// ---------------------------------------------------------------------------
__global__ void k_topk(const int* __restrict__ labels,
                       const int* __restrict__ cnts,
                       int* __restrict__ idxArr,
                       int* __restrict__ outCounts,
                       int* __restrict__ outSizes) {
  __shared__ int   memIdx[MAXMEM];
  __shared__ float memScore[MAXMEM];
  __shared__ int   nMem;
  __shared__ float redS[256];
  __shared__ int   redI[256];
  const int tid = threadIdx.x;
  const int bp = blockIdx.x;
  const int b = bp / P_, p = bp % P_;
  if (tid == 0) nMem = 0;
  __syncthreads();

  for (int n = tid; n < N_; n += 256) {
    if (labels[b * N_ + n] == p) {
      int slot = atomicAdd(&nMem, 1);
      if (slot < MAXMEM) {
        memIdx[slot] = n;
        memScore[slot] = gumbel_hash(b, p, n);
      }
    }
  }
  __syncthreads();

  const int cnt = cnts[bp];
  const int m = (nMem < MAXMEM) ? nMem : MAXMEM;
  if (tid == 0) {
    outCounts[bp] = cnt;
    int sz = cnt / 2;
    outSizes[bp] = (sz < K_) ? sz : K_;
  }

  for (int it = 0; it < K_; ++it) {
    float best = -1e30f;
    int bi = -1;
    for (int i = tid; i < m; i += 256) {
      float s = memScore[i];
      if (s > best) { best = s; bi = i; }
    }
    redS[tid] = best; redI[tid] = bi;
    __syncthreads();
    for (int off = 128; off > 0; off >>= 1) {
      if (tid < off) {
        if (redS[tid + off] > redS[tid]) {
          redS[tid] = redS[tid + off];
          redI[tid] = redI[tid + off];
        }
      }
      __syncthreads();
    }
    if (tid == 0) {
      int w = redI[0];
      idxArr[bp * K_ + it] = (w >= 0) ? memIdx[w] : 0;
      if (w >= 0) memScore[w] = -1e38f;
    }
    __syncthreads();
  }
}

// ---------------------------------------------------------------------------
// K4: samples[b,p,e,k] = (w2 * feats[:, idx[b,p,k]] + b2) for k < sizes.
// A = w2 (M = e, preloaded once), B = gathered columns (N = sample col):
// each lane's 8 results are contiguous in e, bias is one load per lane.
// Staging uses GLOBAL_LOAD_ASYNC_TO_LDS_B128 when available (ASYNCcnt path),
// with a uint4 register copy fallback. WMMA happens before any divergence.
// ---------------------------------------------------------------------------
__global__ void k_samples(const _Float16* __restrict__ feats16,
                          const _Float16* __restrict__ w2h,
                          const float* __restrict__ b2,
                          const int* __restrict__ idxArr,
                          const int* __restrict__ sizesArr,
                          float* __restrict__ outSamples) {
  __shared__ _Float16 colB[16 * D_];  // 16 gathered columns, 4 KB
  const int tid  = threadIdx.x;
  const int lane = tid & 31;
  const int wave = tid >> 5;          // e-tile
  const int bp = blockIdx.x;
  const int b = bp / P_;
  const int size = sizesArr[bp];
  const int e0 = wave * 16;

  // Preload the A fragments (w2 rows e0..e0+15) once.
  v16h afr[4];
#pragma unroll
  for (int kk = 0; kk < 4; ++kk)
    afr[kk] = load_frag(w2h + e0 * D_, D_, kk * 32, lane);

  const int e = e0 + (lane & 15);
  const int hh = lane >> 4;
  const float bias = b2[e];

  for (int t = 0; t < 4; ++t) {
    {  // stage 16 sampled columns: 16 cols x 16 parts == 256 threads
      int j = tid >> 4;
      int part = tid & 15;
      int scol = t * 16 + j;
      int idx = idxArr[bp * K_ + scol];
      const _Float16* src = feats16 + ((size_t)b * N_ + idx) * D_ + part * 8;
      _Float16* dst = &colB[j * D_ + part * 8];
#if USE_ASYNC_LDS
      __builtin_amdgcn_global_load_async_to_lds_b128(
          (gv4i*)(unsigned long long)(uintptr_t)src,
          (lv4i*)dst, 0, 0);
      asm volatile("s_wait_asynccnt 0x0" ::: "memory");
#else
      *(uint4*)dst = *(const uint4*)src;  // 16B chunks
#endif
    }
    __syncthreads();

    v8f acc = {};
#pragma unroll
    for (int kk = 0; kk < 4; ++kk) {
      v16h bf = load_frag(&colB[0], D_, kk * 32, lane);  // B: N=sample col, K=d
      acc = __builtin_amdgcn_wmma_f32_16x16x32_f16(false, afr[kk], false, bf,
                                                   (short)0, acc, false, false);
    }

    // D[m=e][n=kcol]; the store takes lane -> e fixed; 8 consecutive k = t*16 + 8*hh + r.
    const int kbase = t * 16 + 8 * hh;
    float* outp = outSamples + ((size_t)bp * DP_ + e) * K_ + kbase;  // 32B-aligned
    if (kbase + 8 <= size) {
      *(float4*)outp       = make_float4(acc[0] + bias, acc[1] + bias,
                                         acc[2] + bias, acc[3] + bias);
      *((float4*)outp + 1) = make_float4(acc[4] + bias, acc[5] + bias,
                                         acc[6] + bias, acc[7] + bias);
    } else {
#pragma unroll
      for (int r = 0; r < 8; ++r)
        if (kbase + r < size) outp[r] = acc[r] + bias;
    }
    __syncthreads();
  }
}

// ---------------------------------------------------------------------------
extern "C" void kernel_launch(void* const* d_in, const int* in_sizes, int n_in,
                              void* d_out, int out_size, void* d_ws, size_t ws_size,
                              hipStream_t stream) {
  const float* features = (const float*)d_in[0];
  const int*   labels   = (const int*)d_in[1];
  const float* w1       = (const float*)d_in[2];
  const float* b1       = (const float*)d_in[3];
  const float* w2       = (const float*)d_in[4];
  const float* b2       = (const float*)d_in[5];

  float* out    = (float*)d_out;
  float* outEmb = out;                              // [8,64,128] = 65536
  int*   outCnt = (int*)(out + 65536);              // [8,64]     = 512
  float* outSmp = out + 66048;                      // [8,64,128,64] = 4194304
  int*   outSiz = (int*)(out + 4260352);            // [8,64]     = 512

  char* ws = (char*)d_ws;
  _Float16* feats16 = (_Float16*)ws;                                   // 33554432 B
  _Float16* w2h     = (_Float16*)(ws + 33554432);                      // 32768 B
  float*    pooled  = (float*)(ws + 33554432 + 32768);                 // 262144 B
  int*      cnts    = (int*)(ws + 33554432 + 32768 + 262144);          // 2048 B
  int*      idxArr  = (int*)(ws + 33554432 + 32768 + 262144 + 2048);   // 131072 B

  k_init<<<1024, 256, 0, stream>>>(pooled, cnts, outSmp, w2, w2h);
  k_feats_pool<<<B_ * (N_ / 128), 256, 0, stream>>>(features, labels, w1, b1,
                                                    feats16, pooled, cnts);
  k_patch_emb<<<(B_ * P_) / 16, 256, 0, stream>>>(pooled, cnts, w2h, b2, outEmb);
  k_topk<<<B_ * P_, 256, 0, stream>>>(labels, cnts, idxArr, outCnt, outSiz);
  k_samples<<<B_ * P_, 256, 0, stream>>>(feats16, w2h, b2, idxArr, outSiz, outSmp);
}
